// MultiTaskSurgeGNN_10282151707181
// MI455X (gfx1250) — matrix operand order repaired and checked by reading, not compiled, not run on hardware
//
#include <hip/hip_runtime.h>
#include <hip/hip_bf16.h>

// MultiTaskSurgeGNN for MI455X (gfx1250, wave32, WMMA)
// x->bf16 + degree + weight preconvert -> [zero agg, bf16 scatter-mean, WMMA layer] x4 -> WMMA heads
// Activations live in bf16 between layers (WMMA consumes bf16 anyway); agg/atomics stay f32.

#define HDIM    128
#define NLAYERS 4
#define KHEADS  5
#define HHID    64
#define HSTR    321   // padded LDS row stride (321 mod 64 banks = 1 -> conflict-free column walk)

typedef __attribute__((ext_vector_type(16))) __bf16 v16bf;
typedef __attribute__((ext_vector_type(8)))  float  v8f;

struct P32 { uint4 a; uint4 b; };                // 32-byte fragment container

static __device__ __forceinline__ unsigned short f2bf(float f) {
    unsigned int u = __float_as_uint(f);
    u += 0x7FFFu + ((u >> 16) & 1u);             // round-to-nearest-even
    return (unsigned short)(u >> 16);
}
static __device__ __forceinline__ float bf2f(unsigned int us16) {
    return __uint_as_float(us16 << 16);
}

// A fragment (16x32 bf16, M x K) from LDS: lane half holds K = half*16+0..15
static __device__ __forceinline__ v16bf loadA_lds(const unsigned short* s, int row,
                                                  int half, int kc) {
    P32 p;
    p.a = *(const uint4*)(s + row * HDIM + kc + half * 16);
    p.b = *(const uint4*)(s + row * HDIM + kc + half * 16 + 8);
    return __builtin_bit_cast(v16bf, p);
}

// B fragment (32x16 bf16, K x N) from pre-converted bf16 weights in global (L2-hot):
// lane = column, lanes 0-15 hold K=0..15, lanes 16-31 hold K=16..31 -> 32 contiguous bytes
static __device__ __forceinline__ v16bf loadB_bf(const unsigned short* __restrict__ wrow) {
    P32 p;
    p.a = ((const uint4*)wrow)[0];
    p.b = ((const uint4*)wrow)[1];
    return __builtin_bit_cast(v16bf, p);
}

static __device__ __forceinline__ v8f wmma_bf16(v16bf a, v16bf b, v8f c) {
    return __builtin_amdgcn_wmma_f32_16x16x32_bf16(false, a, false, b, (short)0, c,
                                                   false, false);
}

// ---------------------------------------------------------------- utilities
__global__ void zero_f4_kernel(float4* __restrict__ p, int n4) {
    int i = blockIdx.x * blockDim.x + threadIdx.x;
    if (i < n4) p[i] = make_float4(0.f, 0.f, 0.f, 0.f);
}

// f32 -> bf16 pack (weights once per launch; input x once per launch)
__global__ void cvt_bf16_kernel(const float* __restrict__ src,
                                unsigned short* __restrict__ dst, int n4) {
    int i = blockIdx.x * blockDim.x + threadIdx.x;
    if (i >= n4) return;
    float4 v = ((const float4*)src)[i];
    unsigned int lo = (unsigned)f2bf(v.x) | ((unsigned)f2bf(v.y) << 16);
    unsigned int hi = (unsigned)f2bf(v.z) | ((unsigned)f2bf(v.w) << 16);
    ((uint2*)dst)[i] = make_uint2(lo, hi);
}

__global__ void degree_kernel(const int* __restrict__ dst, float* __restrict__ deg, int E) {
    int e = blockIdx.x * blockDim.x + threadIdx.x;
    if (e < E) unsafeAtomicAdd(&deg[dst[e]], 1.0f);
}

// wave-per-edge: 32 lanes x (4 bf16) = contiguous 256B row gather, expand to f32,
// 4 native f32 atomic adds per lane. Edge/node ids are wave-uniform -> SGPRs.
__global__ void scatter_kernel(const unsigned short* __restrict__ xbf,
                               const int* __restrict__ src, const int* __restrict__ dst,
                               float* __restrict__ agg, int E) {
    int t = blockIdx.x * blockDim.x + threadIdx.x;
    int e = __builtin_amdgcn_readfirstlane(t >> 5);
    if (e >= E) return;
    int c = (threadIdx.x & 31) << 2;
    int s = __builtin_amdgcn_readfirstlane(src[e]);
    int d = __builtin_amdgcn_readfirstlane(dst[e]);
    uint2 v = *(const uint2*)(xbf + (size_t)s * HDIM + c);
    float* ap = agg + (size_t)d * HDIM + c;
    unsafeAtomicAdd(ap + 0, bf2f(v.x & 0xFFFFu));
    unsafeAtomicAdd(ap + 1, bf2f(v.x >> 16));
    unsafeAtomicAdd(ap + 2, bf2f(v.y & 0xFFFFu));
    unsafeAtomicAdd(ap + 3, bf2f(v.y >> 16));
}

// ------------------------------------------------- SAGE layer (WMMA bf16)
// x_out = bf16( relu(BN( agg*invdeg @ Wl^T + bl + x_in @ Wr^T )) )
// Block: 256 threads (8 waves), tile 64 nodes x 128 outputs; wave w owns columns w*16..+15.
__global__ __launch_bounds__(256) void sage_layer_kernel(
        const unsigned short* __restrict__ xin, const float* __restrict__ agg,
        const float* __restrict__ deg,
        const unsigned short* __restrict__ Wl, const unsigned short* __restrict__ Wr,
        const float* __restrict__ bl, const float* __restrict__ gma,
        const float* __restrict__ bta, const float* __restrict__ rmean,
        const float* __restrict__ rvar, unsigned short* __restrict__ xout, int n) {
    __shared__ unsigned short sX[64 * HDIM];   // 16 KB bf16 root tile (raw copy)
    __shared__ unsigned short sG[64 * HDIM];   // 16 KB bf16 mean-agg tile

    const int tid   = threadIdx.x;
    const int nbase = blockIdx.x * 64;

    // root tile: straight bf16 copy, 16B per thread-iter
    for (int j = 0; j < 4; ++j) {
        int elem = (tid + j * 256) * 8;         // 64*128 bf16 elements
        int node = nbase + (elem >> 7);
        uint4 v = make_uint4(0u, 0u, 0u, 0u);
        if (node < n) v = *(const uint4*)(xin + (size_t)node * HDIM + (elem & 127));
        *(uint4*)(sX + elem) = v;
    }
    // agg tile: f32 -> (scale by 1/max(deg,1)) -> bf16
    for (int j = 0; j < 8; ++j) {
        int elem = (tid + j * 256) * 4;
        int node = nbase + (elem >> 7);
        float4 vg = make_float4(0.f, 0.f, 0.f, 0.f);
        if (node < n) {
            float inv = 1.0f / fmaxf(deg[node], 1.0f);
            float4 a = *(const float4*)(agg + (size_t)node * HDIM + (elem & 127));
            vg = make_float4(a.x * inv, a.y * inv, a.z * inv, a.w * inv);
        }
        unsigned int lo = (unsigned)f2bf(vg.x) | ((unsigned)f2bf(vg.y) << 16);
        unsigned int hi = (unsigned)f2bf(vg.z) | ((unsigned)f2bf(vg.w) << 16);
        *(uint2*)(sG + elem) = make_uint2(lo, hi);
    }
    __syncthreads();

    const int wave = tid >> 5, lane = tid & 31;
    const int rowh = lane & 15, half = lane >> 4;
    const int o = wave * 16 + rowh;             // this lane's output column

    v8f acc[4] = {};
    const unsigned short* pWl = Wl + (size_t)o * HDIM + half * 16;
    const unsigned short* pWr = Wr + (size_t)o * HDIM + half * 16;
    for (int kc = 0; kc < HDIM; kc += 32) {
        v16bf bfl = loadB_bf(pWl + kc);
        v16bf bfr = loadB_bf(pWr + kc);
        v16bf ag[4], ax[4];
#pragma unroll
        for (int mt = 0; mt < 4; ++mt) {        // batch LDS loads
            ag[mt] = loadA_lds(sG, mt * 16 + rowh, half, kc);
            ax[mt] = loadA_lds(sX, mt * 16 + rowh, half, kc);
        }
#pragma unroll
        for (int mt = 0; mt < 4; ++mt) {        // back-to-back WMMAs
            acc[mt] = wmma_bf16(ag[mt], bfl, acc[mt]);
            acc[mt] = wmma_bf16(ax[mt], bfr, acc[mt]);
        }
    }

    // fused bias + BatchNorm(eval) + ReLU epilogue, bf16 store
    float scale = gma[o] * rsqrtf(rvar[o] + 1e-5f);
    float c0    = (bl[o] - rmean[o]) * scale + bta[o];
#pragma unroll
    for (int mt = 0; mt < 4; ++mt) {
#pragma unroll
        for (int r = 0; r < 8; ++r) {           // C layout: M = r + half*8, N = lane&15
            int node = nbase + mt * 16 + half * 8 + r;
            if (node < n) {
                float v = fmaxf(acc[mt][r] * scale + c0, 0.0f);
                xout[(size_t)node * HDIM + o] = f2bf(v);
            }
        }
    }
}

// ------------------------------------------------- heads (WMMA + LDS reduce)
// h = relu(x @ hW1^T + hb1) as one [N,128]x[128,320] GEMM (o = k*64+oo),
// then per (node,head): sigmoid(<h, hW2[k]> + hb2[k]).
__global__ __launch_bounds__(256) void heads_kernel(
        const unsigned short* __restrict__ x, const unsigned short* __restrict__ W1bf,
        const float* __restrict__ hb1, const float* __restrict__ hW2,
        const float* __restrict__ hb2, float* __restrict__ out, int n) {
    __shared__ unsigned short sX[32 * HDIM];        // 8 KB (raw bf16 copy)
    __shared__ float sH[32 * HSTR];                 // ~41 KB
    __shared__ float sW2[KHEADS * HHID];            // 1.25 KB

    const int tid   = threadIdx.x;
    const int nbase = blockIdx.x * 32;

    for (int j = 0; j < 2; ++j) {
        int elem = (tid + j * 256) * 8;             // 32*128 bf16 elements
        int node = nbase + (elem >> 7);
        uint4 v = make_uint4(0u, 0u, 0u, 0u);
        if (node < n) v = *(const uint4*)(x + (size_t)node * HDIM + (elem & 127));
        *(uint4*)(sX + elem) = v;
    }
    for (int j = tid; j < KHEADS * HHID; j += 256) sW2[j] = hW2[j];
    __syncthreads();

    const int wave = tid >> 5, lane = tid & 31;
    const int rowh = lane & 15, half = lane >> 4;

    for (int t = wave; t < 40; t += 8) {            // 2 M-tiles x 20 N-tiles
        int mt = t & 1, nt = t >> 1;
        int o  = nt * 16 + rowh;                    // 0..319 == k*64+oo
        v8f acc = {};
        const unsigned short* pW = W1bf + (size_t)o * HDIM + half * 16;
        v16bf bf0 = loadB_bf(pW);
        v16bf bf1 = loadB_bf(pW + 32);
        v16bf bf2 = loadB_bf(pW + 64);
        v16bf bf3 = loadB_bf(pW + 96);
        v16bf a0 = loadA_lds(sX, mt * 16 + rowh, half, 0);
        v16bf a1 = loadA_lds(sX, mt * 16 + rowh, half, 32);
        v16bf a2 = loadA_lds(sX, mt * 16 + rowh, half, 64);
        v16bf a3 = loadA_lds(sX, mt * 16 + rowh, half, 96);
        acc = wmma_bf16(a0, bf0, acc);
        acc = wmma_bf16(a1, bf1, acc);
        acc = wmma_bf16(a2, bf2, acc);
        acc = wmma_bf16(a3, bf3, acc);
        float b1 = hb1[o];
#pragma unroll
        for (int r = 0; r < 8; ++r) {
            int ml = mt * 16 + half * 8 + r;
            sH[ml * HSTR + o] = fmaxf(acc[r] + b1, 0.0f);
        }
    }
    __syncthreads();

    if (tid < 32 * KHEADS) {
        int k = tid >> 5, m = tid & 31;
        int node = nbase + m;
        if (node < n) {
            float z = hb2[k];
            const float* hrow = sH + m * HSTR + k * HHID;
            const float* w    = sW2 + k * HHID;
#pragma unroll 8
            for (int oo = 0; oo < HHID; ++oo) z += hrow[oo] * w[oo];
            out[(size_t)k * n + node] = 1.0f / (1.0f + __expf(-z));
        }
    }
}

// ---------------------------------------------------------------- launcher
extern "C" void kernel_launch(void* const* d_in, const int* in_sizes, int n_in,
                              void* d_out, int out_size, void* d_ws, size_t ws_size,
                              hipStream_t stream) {
    const float* x0    = (const float*)d_in[0];
    const int*   ei    = (const int*)  d_in[1];
    const float* Wl    = (const float*)d_in[2];
    const float* blp   = (const float*)d_in[3];
    const float* Wr    = (const float*)d_in[4];
    const float* gma   = (const float*)d_in[5];
    const float* bta   = (const float*)d_in[6];
    const float* rmean = (const float*)d_in[7];
    const float* rvar  = (const float*)d_in[8];
    const float* hW1   = (const float*)d_in[9];
    const float* hb1   = (const float*)d_in[10];
    const float* hW2   = (const float*)d_in[11];
    const float* hb2   = (const float*)d_in[12];
    float* out = (float*)d_out;

    const int Nn = in_sizes[0] / HDIM;
    const int E  = in_sizes[1] / 2;
    const size_t NH = (size_t)Nn * HDIM;
    const int WELEM = NLAYERS * HDIM * HDIM;            // per weight stack (elements)
    const int W1E   = KHEADS * HHID * HDIM;             // head layer-1 weights

    // f32 region first, then 64B-aligned bf16 region
    float* agg = (float*)d_ws;
    float* deg = agg + NH;
    size_t byteoff = ((size_t)((char*)(deg + Nn) - (char*)d_ws) + 63) & ~(size_t)63;
    unsigned short* bx0  = (unsigned short*)((char*)d_ws + byteoff);
    unsigned short* bufA = bx0 + NH;
    unsigned short* bufB = bufA + NH;
    unsigned short* wlbf = bufB + NH;
    unsigned short* wrbf = wlbf + WELEM;
    unsigned short* w1bf = wrbf + WELEM;

    const int* srcp = ei;
    const int* dstp = ei + E;

    // one-time per launch: degree, x -> bf16, weights -> bf16
    {
        int n4 = Nn / 4;
        zero_f4_kernel<<<(n4 + 255) / 256, 256, 0, stream>>>((float4*)deg, n4);
        degree_kernel<<<(E + 255) / 256, 256, 0, stream>>>(dstp, deg, E);
        cvt_bf16_kernel<<<(int)(NH / 4 + 255) / 256, 256, 0, stream>>>(x0, bx0,
                                                                       (int)(NH / 4));
        cvt_bf16_kernel<<<(WELEM / 4 + 255) / 256, 256, 0, stream>>>(Wl, wlbf, WELEM / 4);
        cvt_bf16_kernel<<<(WELEM / 4 + 255) / 256, 256, 0, stream>>>(Wr, wrbf, WELEM / 4);
        cvt_bf16_kernel<<<(W1E / 4 + 255) / 256, 256, 0, stream>>>(hW1, w1bf, W1E / 4);
    }

    const unsigned short* xcur = bx0;
    unsigned short* bufs[2] = {bufA, bufB};
    for (int i = 0; i < NLAYERS; ++i) {
        int n4 = (int)(NH / 4);
        zero_f4_kernel<<<(n4 + 255) / 256, 256, 0, stream>>>((float4*)agg, n4);
        long long sth = (long long)E * 32;
        scatter_kernel<<<(int)((sth + 255) / 256), 256, 0, stream>>>(xcur, srcp, dstp,
                                                                     agg, E);
        unsigned short* xnext = bufs[i & 1];
        sage_layer_kernel<<<(Nn + 63) / 64, 256, 0, stream>>>(
            xcur, agg, deg,
            wlbf + (size_t)i * HDIM * HDIM, wrbf + (size_t)i * HDIM * HDIM,
            blp + (size_t)i * HDIM, gma + (size_t)i * HDIM, bta + (size_t)i * HDIM,
            rmean + (size_t)i * HDIM, rvar + (size_t)i * HDIM, xnext, Nn);
        xcur = xnext;
    }

    heads_kernel<<<(Nn + 31) / 32, 256, 0, stream>>>(xcur, w1bf, hb1, hW2, hb2, out, Nn);
}
